// HeadsSelection_79551384256914
// MI455X (gfx1250) — compile-verified, matrix operand order checked by reading, not run on hardware
//
#include <hip/hip_runtime.h>
#include <hip/hip_bf16.h>

typedef __attribute__((ext_vector_type(16))) __bf16 v16bf;
typedef __attribute__((ext_vector_type(8)))  float  v8f;
typedef __attribute__((ext_vector_type(4)))  unsigned int u32x4;
typedef __attribute__((ext_vector_type(8)))  int i32x8;
typedef __attribute__((ext_vector_type(4)))  int i32x4;

#define S_LEN 256
#define BATCH 16
#define EMB   512
#define HID   256
#define HMDIM 128

// ---------------- fast transcendentals (v_exp_f32 / v_rcp_f32 path) ----------
__device__ __forceinline__ float fast_sig(float x) {
  float e = __builtin_amdgcn_exp2f(-x * 1.442695041f);
  return __builtin_amdgcn_rcpf(1.0f + e);
}
__device__ __forceinline__ float fast_tanh(float x) {
  float e = __builtin_amdgcn_exp2f(x * 2.885390082f);   // exp(2x)
  return 1.0f - 2.0f * __builtin_amdgcn_rcpf(e + 1.0f);
}

// ---------------- WMMA bf16 fragment loader ----------------------------------
// Works for A (row-major [M][K]) and B-as-weights (row-major [N][K]).
// ISA 16-bit layout: lanes 0-15 hold k {0..7,16..23}, lanes 16-31 hold {8..15,24..31}.
__device__ __forceinline__ v16bf load_frag(const __bf16* base, int ldk, int rc, int kbase, int lane) {
  const __bf16* p = base + (size_t)(rc + (lane & 15)) * ldk + kbase + ((lane >> 4) & 1) * 8;
  union { uint4 q[2]; v16bf f; } u;
  u.q[0] = *(const uint4*)p;         // k .. k+7        -> b128 / ds_load_b128
  u.q[1] = *(const uint4*)(p + 16);  // k+16 .. k+23
  return u.f;
}

#define WMMA_BF16(Af, Bf, Cf) \
  __builtin_amdgcn_wmma_f32_16x16x32_bf16(false, (Af), false, (Bf), (short)0, (Cf), false, false)

// ---------------- Tensor Data Mover: 2D bf16 tile -> LDS ----------------------
// D# per CDNA5 ISA 8.3/8.4: group0 = {count/lds_addr/global_addr/type=2},
// group1 = {data_size=2B, tensor_dim0/1, tile_dim0/1, dim0 stride}, groups 2/3
// zero (<=2D tensor). 6-arg builtin form (clang-23 / therock-10.0 headers).
__device__ __forceinline__ void tdm_load_2d(unsigned long long gaddr, unsigned int lds_off,
                                            unsigned int tensor_d0, unsigned int tensor_d1,
                                            unsigned int stride0,
                                            unsigned int tile_d0, unsigned int tile_d1) {
  u32x4 g0;
  g0[0] = 1u;                                                  // count=1 (valid), user mode
  g0[1] = lds_off;                                             // lds_addr (bytes)
  g0[2] = (unsigned int)(gaddr & 0xFFFFFFFFull);               // global_addr[31:0]
  g0[3] = (unsigned int)((gaddr >> 32) & 0x1FFFFFFull) | (2u << 30);  // addr[56:32] | type=2
  i32x8 g1;
  g1[0] = (int)(1u << 16);                                     // data_size=1 -> 2 bytes
  g1[1] = (int)((tensor_d0 & 0xFFFFu) << 16);                  // tensor_dim0[15:0]
  g1[2] = (int)((tensor_d0 >> 16) | ((tensor_d1 & 0xFFFFu) << 16));
  g1[3] = (int)((tensor_d1 >> 16) | (tile_d0 << 16));          // tile_dim0
  g1[4] = (int)(tile_d1 & 0xFFFFu);                            // tile_dim1 (tile_dim2=0)
  g1[5] = (int)stride0;                                        // tensor_dim0_stride[31:0]
  g1[6] = 0;
  g1[7] = 0;
  i32x4 z4; z4[0] = 0; z4[1] = 0; z4[2] = 0; z4[3] = 0;
  i32x8 z8;
#pragma unroll
  for (int i = 0; i < 8; ++i) z8[i] = 0;
  __builtin_amdgcn_tensor_load_to_lds(g0, g1, z4, z4, z8, 0);
}

// ---------------- GEMM: C[M][N] = A[M][K](bf16) x W[N][K]^T(bf16) -------------
// Block = 8 waves covering WM m-tiles x (8/WM) 64-wide n-strips. Wave 0 TDM-stages
// the block's A rows into double-buffered LDS; all waves read A fragments via DS.
// Inner loop is software-pipelined so b128 weight loads overlap the WMMAs.
template<int WM, bool HB>
__global__ __launch_bounds__(256)
void gemm_bf16_tdm(const __bf16* __restrict__ A, const __bf16* __restrict__ W,
                   const float* __restrict__ bias, float* __restrict__ C,
                   int M, int N, int K) {
  constexpr int WN = 8 / WM;
  constexpr int KT = 128;                       // staged k-chunk (elements)
  __shared__ __align__(16) __bf16 aStage[2][WM * 16 * KT];

  int tid    = threadIdx.x;
  int lane   = tid & 31;
  int wave   = tid >> 5;
  int m_base = blockIdx.x * (WM * 16);
  int n_base = blockIdx.y * (WN * 64);
  int mi     = wave / WN;
  int n0     = n_base + (wave % WN) * 64;
  int m0     = m_base + mi * 16;

  v8f acc[4];
#pragma unroll
  for (int g = 0; g < 4; ++g) {
    float bv = HB ? bias[n0 + g * 16 + (lane & 15)] : 0.0f;
#pragma unroll
    for (int r = 0; r < 8; ++r) acc[g][r] = bv;
  }

  if (wave == 0) {
    tdm_load_2d((unsigned long long)(uintptr_t)(A + (size_t)m_base * K),
                (unsigned int)(unsigned long long)(uintptr_t)&aStage[0][0],
                (unsigned int)K, (unsigned int)M, (unsigned int)K,
                (unsigned int)KT, (unsigned int)(WM * 16));
  }

  int stage = 0;
  for (int kb = 0; kb < K; kb += KT, stage ^= 1) {
    if (kb + KT < K) {
      if (wave == 0) {
        tdm_load_2d((unsigned long long)(uintptr_t)(A + (size_t)m_base * K + kb + KT),
                    (unsigned int)(unsigned long long)(uintptr_t)&aStage[stage ^ 1][0],
                    (unsigned int)K, (unsigned int)M, (unsigned int)K,
                    (unsigned int)KT, (unsigned int)(WM * 16));
        __builtin_amdgcn_s_wait_tensorcnt(1);   // current stage landed, next in flight
      }
    } else {
      if (wave == 0) __builtin_amdgcn_s_wait_tensorcnt(0);
    }
    __syncthreads();

    const __bf16* As = &aStage[stage][mi * 16 * KT];
    __builtin_prefetch(W + (size_t)(n0 + (lane & 15)) * K + kb + KT, 0, 1);

    v16bf af  = load_frag(As, KT, 0, 0, lane);
    v16bf bf0 = load_frag(W, K, n0,      kb, lane);
    v16bf bf1 = load_frag(W, K, n0 + 16, kb, lane);
    v16bf bf2 = load_frag(W, K, n0 + 32, kb, lane);
    v16bf bf3 = load_frag(W, K, n0 + 48, kb, lane);
#pragma unroll
    for (int kc = 0; kc < KT; kc += 32) {
      v16bf afn = af, b0n = bf0, b1n = bf1, b2n = bf2, b3n = bf3;
      if (kc + 32 < KT) {                       // prefetch next chunk into fresh regs
        afn = load_frag(As, KT, 0, kc + 32, lane);
        b0n = load_frag(W, K, n0,      kb + kc + 32, lane);
        b1n = load_frag(W, K, n0 + 16, kb + kc + 32, lane);
        b2n = load_frag(W, K, n0 + 32, kb + kc + 32, lane);
        b3n = load_frag(W, K, n0 + 48, kb + kc + 32, lane);
      }
      acc[0] = WMMA_BF16(af, bf0, acc[0]);
      acc[1] = WMMA_BF16(af, bf1, acc[1]);
      acc[2] = WMMA_BF16(af, bf2, acc[2]);
      acc[3] = WMMA_BF16(af, bf3, acc[3]);
      af = afn; bf0 = b0n; bf1 = b1n; bf2 = b2n; bf3 = b3n;
    }
    __syncthreads();
  }

  int col   = lane & 15;
  int rbase = (lane >> 4) ? 8 : 0;
#pragma unroll
  for (int g = 0; g < 4; ++g)
#pragma unroll
    for (int r = 0; r < 8; ++r)
      C[(size_t)(m0 + rbase + r) * N + n0 + g * 16 + col] = acc[g][r];
}

// ---------------- persistent LSTM recurrence ---------------------------------
// One workgroup, 16 waves. Wave w owns gate tiles {i,f,g,o} for hidden j in
// [16w,16w+16): the whole LSTM cell update runs on the wave's own accumulators,
// c stays in 8 VGPRs/lane across all 256 steps, h (bf16) lives in LDS as the
// next step's WMMA A-matrix.
template<int REV>
__global__ __launch_bounds__(512)
void lstm_rec(const float* __restrict__ gx,      // [S*B][1024] = x@w_ih^T + b
              const __bf16* __restrict__ Whh,    // [1024][256]
              __bf16* __restrict__ xout,         // [S*B][512], write cols off..off+255
              int off) {
  __shared__ __align__(16) __bf16 hlds[BATCH * HID];
  int tid   = threadIdx.x;
  int lane  = tid & 31;
  int wave  = tid >> 5;                 // 0..15
  int j     = wave * 16 + (lane & 15);
  int rbase = (lane >> 4) ? 8 : 0;

  for (int i = tid; i < BATCH * HID; i += 512) hlds[i] = (__bf16)0.0f;
  float creg[8];
#pragma unroll
  for (int r = 0; r < 8; ++r) creg[r] = 0.0f;
  __syncthreads();

  for (int step = 0; step < S_LEN; ++step) {
    int s = REV ? (S_LEN - 1 - step) : step;

    v8f acc[4];
#pragma unroll
    for (int g = 0; g < 4; ++g) {
      int n = g * HID + j;
#pragma unroll
      for (int r = 0; r < 8; ++r)
        acc[g][r] = gx[(size_t)(s * BATCH + rbase + r) * 1024 + n];
    }

#pragma unroll
    for (int kb = 0; kb < HID; kb += 32) {
      v16bf af = load_frag(hlds, HID, 0, kb, lane);
#pragma unroll
      for (int g = 0; g < 4; ++g) {
        v16bf bf = load_frag(Whh, HID, g * HID + wave * 16, kb, lane);
        acc[g] = WMMA_BF16(af, bf, acc[g]);
      }
    }
    __syncthreads();   // all reads of hlds done before overwrite

#pragma unroll
    for (int r = 0; r < 8; ++r) {
      float cn = fast_sig(acc[1][r]) * creg[r] + fast_sig(acc[0][r]) * fast_tanh(acc[2][r]);
      creg[r]  = cn;
      float hv = fast_sig(acc[3][r]) * fast_tanh(cn);
      int m = rbase + r;
      __bf16 hb = (__bf16)hv;
      hlds[m * HID + j] = hb;
      xout[(size_t)(s * BATCH + m) * 512 + off + j] = hb;
    }
    __syncthreads();   // h visible before next step's WMMA
  }
}

// ---------------- embedding gather + bf16 cast -------------------------------
__global__ __launch_bounds__(256)
void embed_kernel(const int* __restrict__ concepts, const float* __restrict__ emb,
                  __bf16* __restrict__ x) {
  int row = blockIdx.x;                 // s*B + b
  int cid = concepts[row];
  const float* e = emb + (size_t)cid * EMB;
  __bf16* o = x + (size_t)row * EMB;
  for (int t = threadIdx.x; t < EMB; t += 256) o[t] = (__bf16)e[t];
}

__global__ __launch_bounds__(256)
void cast_bf16(const float* __restrict__ src, __bf16* __restrict__ dst, int n) {
  int i = blockIdx.x * 256 + threadIdx.x;
  if (i < n) dst[i] = (__bf16)src[i];
}

// ---------------- pairwise scorer: va . tanh(p_i + c_j) ----------------------
__global__ __launch_bounds__(256)
void score_kernel(const float* __restrict__ P, const float* __restrict__ Cc,
                  const float* __restrict__ va, float* __restrict__ scores,
                  int* __restrict__ preds) {
  __shared__ float pl[16][HMDIM];
  __shared__ float cl[16][HMDIM];
  __shared__ float vl[HMDIM];
  int b  = blockIdx.z;
  int it = blockIdx.y;
  int jt = blockIdx.x;
  int tid = threadIdx.x;
  if (tid < HMDIM) vl[tid] = va[tid];
  for (int idx = tid; idx < 16 * HMDIM; idx += 256) {
    int rr = idx >> 7, hh = idx & (HMDIM - 1);
    pl[rr][hh] = P [(size_t)((it * 16 + rr) * BATCH + b) * HMDIM + hh];
    cl[rr][hh] = Cc[(size_t)((jt * 16 + rr) * BATCH + b) * HMDIM + hh];
  }
  __syncthreads();
  int ti = tid >> 4, tj = tid & 15;
  float accv = 0.0f;
#pragma unroll 4
  for (int h = 0; h < HMDIM; ++h)
    accv += vl[h] * fast_tanh(pl[ti][h] + cl[tj][h]);
  int i = it * 16 + ti, jd = jt * 16 + tj;
  size_t o = (size_t)b * S_LEN * S_LEN + (size_t)i * S_LEN + jd;
  scores[o] = accv;
  preds[o]  = (accv >= 0.0f) ? 1 : 0;   // sigmoid(x) >= 0.5 <=> x >= 0
}

// ---------------- driver ------------------------------------------------------
extern "C" void kernel_launch(void* const* d_in, const int* in_sizes, int n_in,
                              void* d_out, int out_size, void* d_ws, size_t ws_size,
                              hipStream_t stream) {
  const int*   concepts = (const int*)  d_in[0];
  const float* emb      = (const float*)d_in[2];
  const float* w_ih     = (const float*)d_in[3];   // [2][2][1024][512]
  const float* w_hh     = (const float*)d_in[4];   // [2][2][1024][256]
  const float* bgate    = (const float*)d_in[5];   // [2][2][1024]
  const float* Ua       = (const float*)d_in[6];   // [128][512]
  const float* Wa       = (const float*)d_in[7];   // [128][512]
  const float* va       = (const float*)d_in[8];   // [1][128]

  char* ws = (char*)d_ws;
  __bf16* w_ih_bf = (__bf16*)(ws);                  //  4 MB
  __bf16* w_hh_bf = (__bf16*)(ws +  4194304);       //  2 MB
  __bf16* Ua_bf   = (__bf16*)(ws +  6291456);       //  128 KB
  __bf16* Wa_bf   = (__bf16*)(ws +  6422528);       //  128 KB
  __bf16* xa      = (__bf16*)(ws +  6553600);       //  4 MB  [S*B][512]
  __bf16* xb      = (__bf16*)(ws + 10747904);       //  4 MB
  float*  gx      = (float*) (ws + 14942208);       // 16 MB  [S*B][1024]
  float*  pbuf    = (float*) (ws + 31719424);       //  2 MB  [S*B][128]
  float*  cbuf    = (float*) (ws + 33816576);       //  2 MB

  cast_bf16<<<2097152 / 256, 256, 0, stream>>>(w_ih, w_ih_bf, 2097152);
  cast_bf16<<<1048576 / 256, 256, 0, stream>>>(w_hh, w_hh_bf, 1048576);
  cast_bf16<<<  65536 / 256, 256, 0, stream>>>(Ua, Ua_bf, 65536);
  cast_bf16<<<  65536 / 256, 256, 0, stream>>>(Wa, Wa_bf, 65536);
  embed_kernel<<<S_LEN * BATCH, 256, 0, stream>>>(concepts, emb, xa);

  __bf16* xcur = xa;
  __bf16* xnext = xb;
  for (int l = 0; l < 2; ++l) {
    for (int d = 0; d < 2; ++d) {
      int idx = l * 2 + d;
      // gx = xcur @ w_ih^T + b : M=4096 N=1024 K=512
      // WM=1: block = 1 m-tile x 512 n; grid = (256 m-tiles, 2 n-halves)
      gemm_bf16_tdm<1, true><<<dim3(256, 2), 256, 0, stream>>>(
          xcur, w_ih_bf + (size_t)idx * 1024 * 512, bgate + idx * 1024, gx,
          4096, 1024, 512);
      if (d == 0)
        lstm_rec<0><<<1, 512, 0, stream>>>(gx, w_hh_bf + (size_t)idx * 1024 * 256, xnext, 0);
      else
        lstm_rec<1><<<1, 512, 0, stream>>>(gx, w_hh_bf + (size_t)idx * 1024 * 256, xnext, 256);
    }
    __bf16* t = xcur; xcur = xnext; xnext = t;
  }

  // attention projections: M=4096 N=128 K=512
  // WM=4: block = 4 m-tiles x 128 n; grid = (64, 1)
  gemm_bf16_tdm<4, false><<<dim3(64, 1), 256, 0, stream>>>(
      xcur, Ua_bf, nullptr, pbuf, 4096, 128, 512);
  gemm_bf16_tdm<4, false><<<dim3(64, 1), 256, 0, stream>>>(
      xcur, Wa_bf, nullptr, cbuf, 4096, 128, 512);

  float* scores = (float*)d_out;
  int*   preds  = (int*)d_out + 1048576;      // B*S*S
  dim3 g(16, 16, 16);
  score_kernel<<<g, 256, 0, stream>>>(pbuf, cbuf, va, scores, preds);
}
